// ConstraintAwareBiasing_84559316123933
// MI455X (gfx1250) — matrix operand order repaired and checked by reading, not compiled, not run on hardware
//
#include <hip/hip_runtime.h>
#include <stdint.h>

#define S_LEN 512
#define D_DIM 128
#define H_DIM 128
#define NH 16
#define WAVES_PER_BLOCK 4
#define N_BLOCKS 1024
#define TILES_PER_WAVE 8   // 1024 blocks * 4 waves * 8 = 32768 tiles

typedef __attribute__((ext_vector_type(16))) __bf16 v16bf;
typedef __attribute__((ext_vector_type(8)))  float  v8f;

union FragB16 {
  uint4 q[2];
  v16bf v;
};

__device__ __forceinline__ unsigned int bf16_bits(float f) {
  // round-to-nearest-even bf16 (software fallback)
  unsigned int u = __float_as_uint(f);
  u += 0x7FFFu + ((u >> 16) & 1u);
  return u >> 16;
}
__device__ __forceinline__ unsigned int pack_bf16x2_sw(float lo, float hi) {
  return bf16_bits(lo) | (bf16_bits(hi) << 16);
}

#if __has_builtin(__builtin_amdgcn_cvt_pk_bf16_f32)
#define HAVE_CVT_PK_BF16 1
typedef __attribute__((ext_vector_type(2))) __bf16 v2bf;
__device__ __forceinline__ unsigned int pack_bf16x2(float lo, float hi) {
  union { v2bf v; unsigned int u; } cvt;
  cvt.v = __builtin_amdgcn_cvt_pk_bf16_f32(lo, hi);
  return cvt.u;
}
#else
#define HAVE_CVT_PK_BF16 0
__device__ __forceinline__ unsigned int pack_bf16x2(float lo, float hi) {
  return pack_bf16x2_sw(lo, hi);
}
#endif

__device__ __forceinline__ unsigned short bf16_one(float v) {
#if HAVE_CVT_PK_BF16
  return (unsigned short)(pack_bf16x2(v, v) & 0xffffu);  // b16 store keeps [15:0]
#else
  return (unsigned short)bf16_bits(v);
#endif
}

#if __has_builtin(__builtin_amdgcn_global_load_async_to_lds_b128)
#define HAVE_ASYNC_LDS 1
// Types matching the builtin prototype: int4 in global (AS1) / LDS (AS3).
typedef int vi4 __attribute__((vector_size(16)));
typedef __attribute__((address_space(1))) vi4 gvi4;
typedef __attribute__((address_space(3))) vi4 lvi4;
#else
#define HAVE_ASYNC_LDS 0
#endif

// ---------------------------------------------------------------------------
// Kernel 1: hi[b,s,:] = E[b,s,:] @ W1[0:128,:] + b1 ;  hj = E @ W1[128:256,:]
// ---------------------------------------------------------------------------
__global__ __launch_bounds__(128) void k_hij(const float* __restrict__ E,
                                             const float* __restrict__ W1,
                                             const float* __restrict__ b1,
                                             float* __restrict__ hi,
                                             float* __restrict__ hj) {
  int bs = blockIdx.x;      // 0 .. B*S-1
  int t  = threadIdx.x;     // 0 .. 127
  __shared__ float e[D_DIM];
  e[t] = E[(size_t)bs * D_DIM + t];
  __syncthreads();
  float a1 = 0.f, a2 = 0.f;
#pragma unroll 8
  for (int d = 0; d < D_DIM; ++d) {
    float ev = e[d];
    a1 += ev * W1[(size_t)d * H_DIM + t];
    a2 += ev * W1[(size_t)(D_DIM + d) * H_DIM + t];
  }
  hi[(size_t)bs * H_DIM + t] = a1 + b1[t];
  hj[(size_t)bs * H_DIM + t] = a2;
}

// ---------------------------------------------------------------------------
// Kernel 1b: one-time bf16 conversion of W2^T and W3^T into workspace.
// w2bf[n*128+h] = bf16(W2[h,n]); w3bf[p*128+h] = bf16(W3[h,p])
// ---------------------------------------------------------------------------
__global__ __launch_bounds__(256) void k_wconv(const float* __restrict__ W2,
                                               const float* __restrict__ W3,
                                               unsigned short* __restrict__ w2bf,
                                               unsigned short* __restrict__ w3bf) {
  int tid = blockIdx.x * 256 + threadIdx.x;   // 0 .. 18431
  if (tid < H_DIM * H_DIM) {
    int h = tid >> 7, n = tid & 127;
    w2bf[n * H_DIM + h] = bf16_one(W2[tid]);
  } else {
    int t = tid - H_DIM * H_DIM;
    int h = t >> 4, p = t & 15;
    w3bf[p * H_DIM + h] = bf16_one(W3[t]);
  }
}

// ---------------------------------------------------------------------------
// Kernel 2: fused pairwise MLP + transpose + mask. One wave = one 16-row tile
// (fixed b,i; 16 consecutive j); 8 tiles per wave to amortize weight staging.
// ---------------------------------------------------------------------------
__global__ __launch_bounds__(128) void k_bias(
    const float* __restrict__ hi, const float* __restrict__ hj,
    const unsigned short* __restrict__ w2bf, const unsigned short* __restrict__ w3bf,
    const float* __restrict__ b2, const float* __restrict__ b3,
    const float* __restrict__ temps, const unsigned char* __restrict__ mask,
    float* __restrict__ out) {
  __shared__ __align__(16) unsigned short w2t[H_DIM * H_DIM];                 // 32 KB
  __shared__ __align__(16) unsigned short w3t[NH * H_DIM];                    //  4 KB
  __shared__ __align__(16) unsigned short hbuf[WAVES_PER_BLOCK][16 * H_DIM];  // 4x4 KB

  const int tid = threadIdx.x;

  // ---- stage pre-converted bf16 weights into LDS via async DMA path ----
#if HAVE_ASYNC_LDS
  {
#pragma unroll
    for (int c = 0; c < 16; ++c) {
      int elem = (tid + c * 128) * 8;  // 8 bf16 per b128 chunk
      __builtin_amdgcn_global_load_async_to_lds_b128(
          (gvi4*)(w2bf + elem), (lvi4*)(w2t + elem), 0, 0);
    }
#pragma unroll
    for (int c = 0; c < 2; ++c) {
      int elem = (tid + c * 128) * 8;
      __builtin_amdgcn_global_load_async_to_lds_b128(
          (gvi4*)(w3bf + elem), (lvi4*)(w3t + elem), 0, 0);
    }
#if __has_builtin(__builtin_amdgcn_s_wait_asynccnt)
    __builtin_amdgcn_s_wait_asynccnt(0);
#else
    asm volatile("s_wait_asynccnt 0" ::: "memory");
#endif
  }
#else
  for (int c = tid; c < (H_DIM * H_DIM) / 8; c += 128)
    ((uint4*)w2t)[c] = ((const uint4*)w2bf)[c];
  for (int c = tid; c < (NH * H_DIM) / 8; c += 128)
    ((uint4*)w3t)[c] = ((const uint4*)w3bf)[c];
#endif
  __syncthreads();

  const int wave = tid >> 5;
  const int lane = tid & 31;
  const int lm   = lane & 15;   // N index inside 16-wide tile
  const int half = lane >> 4;   // lane half selects K sub-range (ISA 7.12.2)
  unsigned short* hl = hbuf[wave];  // wave-private: same-wave DS is in-order

  for (int t = 0; t < TILES_PER_WAVE; ++t) {
    const int tileId =
        (blockIdx.x * WAVES_PER_BLOCK + wave) + t * (N_BLOCKS * WAVES_PER_BLOCK);
    const int b   = tileId >> 14;          // / (512*32)
    const int rem = tileId & 16383;
    const int i   = rem >> 5;
    const int j0  = (rem & 31) << 4;

    // ---- build h tile: h[r][k] = relu(hi[b,i,k] + hj[b,j0+r,k]), bf16 ----
    const float4 hi4 =
        *(const float4*)(hi + ((size_t)(b * S_LEN + i)) * H_DIM + lane * 4);
#pragma unroll
    for (int r = 0; r < 16; ++r) {
      float4 hj4 =
          *(const float4*)(hj + ((size_t)(b * S_LEN + j0 + r)) * H_DIM + lane * 4);
      float x0 = fmaxf(hi4.x + hj4.x, 0.f);
      float x1 = fmaxf(hi4.y + hj4.y, 0.f);
      float x2 = fmaxf(hi4.z + hj4.z, 0.f);
      float x3 = fmaxf(hi4.w + hj4.w, 0.f);
      uint2 pk = make_uint2(pack_bf16x2(x0, x1), pack_bf16x2(x2, x3));
      *(uint2*)(hl + r * H_DIM + lane * 4) = pk;
    }

    // ---- layer 2: 8 N-tiles x 4 K-chunks of v_wmma_f32_16x16x32_bf16 ----
    v8f acc[8];
#pragma unroll
    for (int nt = 0; nt < 8; ++nt) {
      v8f z = {0.f, 0.f, 0.f, 0.f, 0.f, 0.f, 0.f, 0.f};
      acc[nt] = z;
    }

#pragma unroll
    for (int kc = 0; kc < 4; ++kc) {
      // A fragment: K = kc*32 + half*16 + {0..15} (contiguous)
      FragB16 A;
      A.q[0] = *(const uint4*)(hl + lm * H_DIM + kc * 32 + half * 16);
      A.q[1] = *(const uint4*)(hl + lm * H_DIM + kc * 32 + half * 16 + 8);
#pragma unroll
      for (int nt = 0; nt < 8; ++nt) {
        // B fragment: N = nt*16+lm, K = kc*32 + half*16 + {0..15} (contiguous)
        FragB16 Bf;
        const uint4* bp =
            (const uint4*)(w2t + (nt * 16 + lm) * H_DIM + kc * 32 + half * 16);
        Bf.q[0] = bp[0];
        Bf.q[1] = bp[1];
        acc[nt] = __builtin_amdgcn_wmma_f32_16x16x32_bf16(
            false, A.v, false, Bf.v, (short)0, acc[nt], false, false);
      }
    }

    // ---- relu(+b2); repack h2 as bf16 row-major into the same LDS tile ----
    // D layout: lane -> N = lm, VGPR g -> M = g + half*8   (ISA 7.12.2)
#pragma unroll
    for (int nt = 0; nt < 8; ++nt) {
      float bb = b2[nt * 16 + lm];
#pragma unroll
      for (int g = 0; g < 8; ++g) {
        float v = fmaxf(acc[nt][g] + bb, 0.f);
        hl[(g + half * 8) * H_DIM + nt * 16 + lm] = bf16_one(v);
      }
    }

    // ---- layer 3: 4 K-chunks against W3^T ----
    v8f dacc = {0.f, 0.f, 0.f, 0.f, 0.f, 0.f, 0.f, 0.f};
#pragma unroll
    for (int kc = 0; kc < 4; ++kc) {
      FragB16 A;
      A.q[0] = *(const uint4*)(hl + lm * H_DIM + kc * 32 + half * 16);
      A.q[1] = *(const uint4*)(hl + lm * H_DIM + kc * 32 + half * 16 + 8);
      FragB16 Bw;
      const uint4* bp = (const uint4*)(w3t + lm * H_DIM + kc * 32 + half * 16);
      Bw.q[0] = bp[0];
      Bw.q[1] = bp[1];
      dacc = __builtin_amdgcn_wmma_f32_16x16x32_bf16(
          false, A.v, false, Bw.v, (short)0, dacc, false, false);
    }

    // ---- epilogue: +b3, *temperature, transpose D tile through LDS ----
    float* dl = (float*)hl;  // reuse wave-private tile (all bf16 reads done)
    {
      float bb = b3[lm];
      float tv = temps[lm];
#pragma unroll
      for (int g = 0; g < 8; g += 2) {
        // dl layout [p][r] : p = lm (head), r = g + half*8 (j offset)
        float2 two = make_float2((dacc[g] + bb) * tv, (dacc[g + 1] + bb) * tv);
        *(float2*)(dl + lm * 16 + half * 8 + g) = two;
      }
    }

    // ---- coalesced masked store: lane -> (head p, 8-wide j chunk c) ----
    const int p = lane >> 1;
    const int c = lane & 1;
    float4 v0 = *(const float4*)(dl + p * 16 + c * 8);
    float4 v1 = *(const float4*)(dl + p * 16 + c * 8 + 4);
    const unsigned char* mrow =
        mask + ((size_t)b * S_LEN + i) * S_LEN + j0 + c * 8;
    const float NI = -__builtin_inff();
    v0.x = mrow[0] ? v0.x : NI;
    v0.y = mrow[1] ? v0.y : NI;
    v0.z = mrow[2] ? v0.z : NI;
    v0.w = mrow[3] ? v0.w : NI;
    v1.x = mrow[4] ? v1.x : NI;
    v1.y = mrow[5] ? v1.y : NI;
    v1.z = mrow[6] ? v1.z : NI;
    v1.w = mrow[7] ? v1.w : NI;
    float* orow =
        out + (((size_t)(b * NH + p) * S_LEN) + i) * S_LEN + j0 + c * 8;
    *(float4*)orow       = v0;
    *(float4*)(orow + 4) = v1;
  }
}

extern "C" void kernel_launch(void* const* d_in, const int* in_sizes, int n_in,
                              void* d_out, int out_size, void* d_ws, size_t ws_size,
                              hipStream_t stream) {
  const float*         E     = (const float*)d_in[0];
  const unsigned char* mask  = (const unsigned char*)d_in[1];  // bool -> byte
  const float*         W1    = (const float*)d_in[2];
  const float*         b1    = (const float*)d_in[3];
  const float*         W2    = (const float*)d_in[4];
  const float*         b2    = (const float*)d_in[5];
  const float*         W3    = (const float*)d_in[6];
  const float*         b3    = (const float*)d_in[7];
  const float*         temps = (const float*)d_in[8];
  float* out = (float*)d_out;

  // workspace: hi | hj | w2bf | w3bf
  float* hi = (float*)d_ws;
  float* hj = hi + 2 * S_LEN * H_DIM;
  unsigned short* w2bf = (unsigned short*)((char*)d_ws + (1 << 20));
  unsigned short* w3bf = w2bf + H_DIM * H_DIM;

  k_hij<<<2 * S_LEN, 128, 0, stream>>>(E, W1, b1, hi, hj);
  k_wconv<<<(H_DIM * H_DIM + NH * H_DIM) / 256, 256, 0, stream>>>(W2, W3, w2bf, w3bf);

  k_bias<<<N_BLOCKS, WAVES_PER_BLOCK * 32, 0, stream>>>(
      hi, hj, w2bf, w3bf, b2, b3, temps, mask, out);
}
